// LSTM_4329327034914
// MI455X (gfx1250) — matrix operand order repaired and checked by reading, not compiled, not run on hardware
//
#include <hip/hip_runtime.h>
#include <hip/hip_bf16.h>
#include <math.h>

#define TT 64
#define BB 256
#define FF 1024
#define KCH 256                 // K-chunk staged in LDS
#define LDSROW (KCH + 8)        // +8 halfs pad: row stride 132 dwords == 4 mod 64 banks
#define BM 32                   // rows per block (2 WMMA M-subtiles per wave)
#define BN 64                   // cols per block (4 waves x 16 cols)
#define NCH (FF / KCH)          // 4 chunks
#define NST (KCH / 32)          // 8 K-steps per chunk

typedef __attribute__((ext_vector_type(16))) __bf16 v16bf;
typedef __attribute__((ext_vector_type(8)))  __bf16 v8bf;
typedef __attribute__((ext_vector_type(8)))  float  v8f;

__device__ __forceinline__ float sigm(float x) { return 1.0f / (1.0f + __expf(-x)); }

// ---------------------------------------------------------------------------
// Convert 8 gate weight matrices [C=2][F][F] fp32 -> bf16:
// dst[(c*4 + g) * F*F + i], g in {f, is, it, os}.
// ---------------------------------------------------------------------------
__global__ void convert_weights(const float* __restrict__ Wf,
                                const float* __restrict__ Wis,
                                const float* __restrict__ Wit,
                                const float* __restrict__ Wos,
                                __bf16* __restrict__ dst) {
  size_t tid = (size_t)blockIdx.x * blockDim.x + threadIdx.x;
  const size_t per = (size_t)2 * FF * FF;
  if (tid >= 4 * per) return;
  int g = (int)(tid / per);
  size_t rem = tid % per;
  const float* src = (g == 0) ? Wf : (g == 1) ? Wis : (g == 2) ? Wit : Wos;
  size_t c = rem / ((size_t)FF * FF);
  size_t i = rem % ((size_t)FF * FF);
  dst[(c * 4 + (size_t)g) * (size_t)FF * FF + i] = (__bf16)src[rem];
}

__global__ void zero_f32(float* __restrict__ p, size_t n) {
  size_t tid = (size_t)blockIdx.x * blockDim.x + threadIdx.x;
  if (tid < n) p[tid] = 0.0f;
}

__global__ void cvt_bf16(const float* __restrict__ src, __bf16* __restrict__ dst, size_t n) {
  size_t tid = (size_t)blockIdx.x * blockDim.x + threadIdx.x;
  if (tid < n) dst[tid] = (__bf16)src[tid];
}

// ---------------------------------------------------------------------------
// One LSTM stage (timestep t, cell c):
//   pre_g = z @ Wg.T + bg  for g in {f, is, it, os}  (4 fused WMMA bf16 GEMMs)
//   f = sig(pre_f); i = sig(pre_is)*tanh(pre_it); o = sig(pre_os)
//   h_new  = tanh(cs_old) * o        (pre-update cell state, per reference)
//   cs_new = cs_old * f + i
//   z_next = bf16(x_next + h_new)    (ping-pong buffer)
//
// Block = 128 threads (4 wave32s) -> 32x64 tile; wave w owns cols
// [n0+16w, n0+16w+16) for all four gates (wave-local pointwise epilogue).
// z chunks are LDS double-buffered (fill of chunk c+1 overlaps compute of
// chunk c; one barrier per chunk). B fragments are software-pipelined one
// K-step ahead so WMMAs are not stalled on weight loads.
// ---------------------------------------------------------------------------
__global__ __launch_bounds__(128) void lstm_stage(
    const __bf16* __restrict__ z,       // [B,F] bf16 (z = x_t + h)
    const __bf16* __restrict__ wcell,   // 4 gate matrices, each [F,F] bf16
    const float* __restrict__ bfv, const float* __restrict__ bisv,
    const float* __restrict__ bitv, const float* __restrict__ bosv,
    float* __restrict__ h,              // [B,F] fp32 (out)
    float* __restrict__ cs,             // [B,F] fp32 for this cell (in/out)
    const float* __restrict__ xnext,    // next stage's x_t (or null)
    __bf16* __restrict__ znext,         // other ping-pong z buffer
    int has_next) {
  __shared__ __bf16 zs[2][BM * LDSROW];  // ~33 KB total

  const int tid  = threadIdx.x;
  const int lane = tid & 31;
  const int w    = tid >> 5;             // wave id 0..3
  const int m0   = blockIdx.y * BM;
  const int n0   = blockIdx.x * BN;

  // B fragment (32x16 K x N, 16-bit): lane L<16 -> K=k0..k0+15 of col n0+16w+L,
  // lanes 16..31 -> K=k0+16..k0+31 (16 contiguous halfs = one 32B load).
  const int nB    = n0 + 16 * w + (lane & 15);
  const int kBoff = (lane < 16) ? 0 : 16;
  const __bf16* wrow[4];
#pragma unroll
  for (int g = 0; g < 4; ++g)
    wrow[g] = wcell + (size_t)g * FF * FF + (size_t)nB * FF + kBoff;

  // A fragment (16x32 M x K, 16-bit): lane L<16 holds K {0..15},
  // lanes 16..31 hold K {16..31}; M = lane%16.
  const int mAbase = lane & 15;
  const int kAoff  = (lane < 16) ? 0 : 16;

  // LDS fill mapping: 32 rows x 512B/chunk-row, 128 threads x 128B each.
  const int rowL = tid >> 2;             // 0..31
  const int segL = (tid & 3) * 64;       // half offset within chunk row
  const __bf16* zsrc = z + (size_t)(m0 + rowL) * FF + segL;
  __bf16* zdstbase = &zs[0][rowL * LDSROW + segL];
  const int bufstride = BM * LDSROW;

  v8f acc[2][4] = {};                    // [mi][gate]
  v16bf bbuf[2][4];                      // pipelined B fragments

  // Prologue: fill chunk 0, preload B for step 0.
#pragma unroll
  for (int j = 0; j < 4; ++j)
    *(v16bf*)(zdstbase + 16 * j) = *(const v16bf*)(zsrc + 16 * j);
#pragma unroll
  for (int g = 0; g < 4; ++g) bbuf[0][g] = *(const v16bf*)(wrow[g]);
  __syncthreads();

#pragma unroll
  for (int c = 0; c < NCH; ++c) {
    // Kick off fill of chunk c+1 into the other LDS buffer (overlaps compute).
    if (c + 1 < NCH) {
#pragma unroll
      for (int j = 0; j < 4; ++j)
        *(v16bf*)(zdstbase + ((c + 1) & 1) * bufstride + 16 * j) =
            *(const v16bf*)(zsrc + (c + 1) * KCH + 16 * j);
    }

#pragma unroll
    for (int s8 = 0; s8 < NST; ++s8) {
      const int step = c * NST + s8;     // compile-time (both loops unrolled)
      const int k0   = step * 32;
      const int cur  = step & 1;

      // Pipeline: issue next step's B loads + prefetch two steps ahead.
      if (step + 1 < NCH * NST) {
#pragma unroll
        for (int g = 0; g < 4; ++g)
          bbuf[cur ^ 1][g] = *(const v16bf*)(wrow[g] + k0 + 32);
        if (k0 + 64 < FF) {
#pragma unroll
          for (int g = 0; g < 4; ++g)
            __builtin_prefetch(wrow[g] + k0 + 64, 0, 1);
        }
      }

      v16bf a[2];
#pragma unroll
      for (int mi = 0; mi < 2; ++mi) {
        const __bf16* ap =
            &zs[c & 1][(16 * mi + mAbase) * LDSROW + s8 * 32 + kAoff];
        v8bf alo = *(const v8bf*)ap;
        v8bf ahi = *(const v8bf*)(ap + 8);
        a[mi] = __builtin_shufflevector(alo, ahi, 0, 1, 2, 3, 4, 5, 6, 7, 8, 9,
                                        10, 11, 12, 13, 14, 15);
      }

#pragma unroll
      for (int mi = 0; mi < 2; ++mi)
#pragma unroll
        for (int g = 0; g < 4; ++g)
          acc[mi][g] = __builtin_amdgcn_wmma_f32_16x16x32_bf16(
              false, a[mi], false, bbuf[cur][g], (short)0, acc[mi][g], false,
              false);
    }

    if (c + 1 < NCH) __syncthreads();    // fill of chunk c+1 complete everywhere
  }

  // Epilogue: D layout -> VGPR r, lane L: (M = 16*mi + r + (L<16?0:8), N = L%16)
  const int n    = n0 + 16 * w + (lane & 15);
  const int mrow = (lane < 16) ? 0 : 8;
  const float bF = bfv[n], bIS = bisv[n], bIT = bitv[n], bOS = bosv[n];

#pragma unroll
  for (int mi = 0; mi < 2; ++mi) {
#pragma unroll
    for (int r = 0; r < 8; ++r) {
      int m = m0 + 16 * mi + r + mrow;
      size_t idx = (size_t)m * FF + n;
      float fg = sigm(acc[mi][0][r] + bF);
      float ig = sigm(acc[mi][1][r] + bIS) * tanhf(acc[mi][2][r] + bIT);
      float og = sigm(acc[mi][3][r] + bOS);
      float cold = cs[idx];
      float hnew = tanhf(cold) * og;
      cs[idx] = cold * fg + ig;
      h[idx]  = hnew;
      if (has_next) znext[idx] = (__bf16)(xnext[idx] + hnew);
    }
  }
}

// ---------------------------------------------------------------------------
// MLP head (~0.9 GFLOP total, plain fp32 — negligible vs the scan)
// ---------------------------------------------------------------------------
__global__ void gemm_bias(const float* __restrict__ X, const float* __restrict__ W,
                          const float* __restrict__ b, float* __restrict__ Y,
                          int N, int K) {
  int t = blockIdx.x * blockDim.x + threadIdx.x;  // t = m*N + n, M = 256
  int m = t / N, n = t % N;
  const float* x  = X + (size_t)m * K;
  const float* wr = W + (size_t)n * K;
  float s = 0.0f;
  for (int k = 0; k < K; ++k) s += x[k] * wr[k];
  Y[t] = s + b[n];
}

__global__ __launch_bounds__(256) void bn_relu(float* __restrict__ X,
                                               const float* __restrict__ g,
                                               const float* __restrict__ be, int N) {
  __shared__ float s1[256], s2[256];
  int n = blockIdx.x;
  int t = threadIdx.x;  // row (B = 256)
  float v = X[(size_t)t * N + n];
  s1[t] = v;
  s2[t] = v * v;
  __syncthreads();
  for (int off = 128; off > 0; off >>= 1) {
    if (t < off) { s1[t] += s1[t + off]; s2[t] += s2[t + off]; }
    __syncthreads();
  }
  float mean = s1[0] * (1.0f / 256.0f);
  float var  = s2[0] * (1.0f / 256.0f) - mean * mean;  // biased var (ddof=0)
  float y = g[n] * (v - mean) * rsqrtf(var + 1e-5f) + be[n];
  X[(size_t)t * N + n] = fmaxf(y, 0.0f);
}

__global__ void head_out(const float* __restrict__ x3, const float* __restrict__ sent,
                         const float* __restrict__ Wout, const float* __restrict__ bout,
                         float* __restrict__ out) {
  int t = blockIdx.x * blockDim.x + threadIdx.x;  // 0..511 : m*2 + o
  int m = t >> 1, o = t & 1;
  const float* wr = Wout + (size_t)o * 257;
  const float* x  = x3 + (size_t)m * 256;
  float s = 0.0f;
  for (int k = 0; k < 256; ++k) s += x[k] * wr[k];
  s += sent[m] * wr[256];
  out[t] = s + bout[o];
}

// ---------------------------------------------------------------------------
extern "C" void kernel_launch(void* const* d_in, const int* in_sizes, int n_in,
                              void* d_out, int out_size, void* d_ws, size_t ws_size,
                              hipStream_t stream) {
  (void)in_sizes; (void)n_in; (void)out_size; (void)ws_size;
  const float* inputs = (const float*)d_in[0];
  const float* sent   = (const float*)d_in[1];
  const float* Wf  = (const float*)d_in[2];
  const float* bfv = (const float*)d_in[3];
  const float* Wis = (const float*)d_in[4];
  const float* bis = (const float*)d_in[5];
  const float* Wit = (const float*)d_in[6];
  const float* bit = (const float*)d_in[7];
  const float* Wos = (const float*)d_in[8];
  const float* bos = (const float*)d_in[9];
  const float* W1 = (const float*)d_in[10]; const float* b1 = (const float*)d_in[11];
  const float* g1 = (const float*)d_in[12]; const float* be1 = (const float*)d_in[13];
  const float* W2 = (const float*)d_in[14]; const float* b2 = (const float*)d_in[15];
  const float* g2 = (const float*)d_in[16]; const float* be2 = (const float*)d_in[17];
  const float* W3 = (const float*)d_in[18]; const float* b3 = (const float*)d_in[19];
  const float* g3 = (const float*)d_in[20]; const float* be3 = (const float*)d_in[21];
  const float* Wout = (const float*)d_in[22]; const float* bout = (const float*)d_in[23];
  float* out = (float*)d_out;

  // Workspace layout (~21.75 MB)
  char* base = (char*)d_ws;
  __bf16* wbf = (__bf16*)base;                                  // 16 MB bf16 weights
  size_t off = (size_t)8 * FF * FF * 2;
  __bf16* zb[2];
  zb[0] = (__bf16*)(base + off); off += (size_t)BB * FF * 2;    // z ping
  zb[1] = (__bf16*)(base + off); off += (size_t)BB * FF * 2;    // z pong
  float* hbuf  = (float*)(base + off); off += (size_t)BB * FF * 4;       // h
  float* csbuf = (float*)(base + off); off += (size_t)2 * BB * FF * 4;   // cs[2]
  float* x1 = (float*)(base + off); off += (size_t)BB * FF * 4;
  float* x2 = (float*)(base + off); off += (size_t)BB * (FF / 2) * 4;
  float* x3 = (float*)(base + off); off += (size_t)BB * (FF / 4) * 4;

  // 1) Weights -> bf16
  {
    size_t n = (size_t)8 * FF * FF;
    convert_weights<<<(unsigned)((n + 255) / 256), 256, 0, stream>>>(Wf, Wis, Wit, Wos, wbf);
  }
  // 2) h = 0, cs = 0 (contiguous)
  {
    size_t n = (size_t)3 * BB * FF;
    zero_f32<<<(unsigned)((n + 255) / 256), 256, 0, stream>>>(hbuf, n);
  }
  // 3) z_0 = bf16(x_0 + h0) = bf16(x_0)
  {
    size_t n = (size_t)BB * FF;
    cvt_bf16<<<(unsigned)((n + 255) / 256), 256, 0, stream>>>(inputs, zb[0], n);
  }

  // 4) 128 sequential stages, ping-pong z buffers
  dim3 sgrid(FF / BN, BB / BM);  // 16 x 8 = 128 blocks, 128 threads each
  int s = 0;
  for (int t = 0; t < TT; ++t) {
    for (int c = 0; c < 2; ++c, ++s) {
      const float* xnext = nullptr;
      int hn = 0;
      if (c == 0)          { xnext = inputs + (size_t)t * BB * FF;       hn = 1; }
      else if (t + 1 < TT) { xnext = inputs + (size_t)(t + 1) * BB * FF; hn = 1; }
      lstm_stage<<<sgrid, 128, 0, stream>>>(
          zb[s & 1], wbf + (size_t)c * 4 * FF * FF,
          bfv + (size_t)c * FF, bis + (size_t)c * FF,
          bit + (size_t)c * FF, bos + (size_t)c * FF,
          hbuf, csbuf + (size_t)c * BB * FF,
          xnext, zb[(s + 1) & 1], hn);
    }
  }

  // 5) MLP head: F -> F -> F/2 -> F/4, each Linear + BN(train) + ReLU
  gemm_bias<<<(BB * FF) / 256, 256, 0, stream>>>(hbuf, W1, b1, x1, FF, FF);
  bn_relu<<<FF, 256, 0, stream>>>(x1, g1, be1, FF);
  gemm_bias<<<(BB * (FF / 2)) / 256, 256, 0, stream>>>(x1, W2, b2, x2, FF / 2, FF);
  bn_relu<<<FF / 2, 256, 0, stream>>>(x2, g2, be2, FF / 2);
  gemm_bias<<<(BB * (FF / 4)) / 256, 256, 0, stream>>>(x2, W3, b3, x3, FF / 4, FF / 2);
  bn_relu<<<FF / 4, 256, 0, stream>>>(x3, g3, be3, FF / 4);

  // 6) concat(x3, sentiment) @ Wout.T + bout -> [B, 2]
  head_out<<<2, 256, 0, stream>>>(x3, sent, Wout, bout, out);
}
